// Attention_2in_3633542332822
// MI455X (gfx1250) — matrix operand order repaired and checked by reading, not compiled, not run on hardware
//
#include <hip/hip_runtime.h>
#include <hip/hip_bf16.h>

// ---------------------------------------------------------------------------
// CDNA5 (gfx1250, wave32) WMMA types and helpers
// ---------------------------------------------------------------------------
typedef __bf16 v16bf __attribute__((ext_vector_type(16)));
typedef __bf16 v2bf  __attribute__((ext_vector_type(2)));
typedef float  v8f   __attribute__((ext_vector_type(8)));
typedef int    v4i   __attribute__((ext_vector_type(4)));

union FragBF { v16bf v; unsigned u[8]; unsigned short h[16]; };
union FragF  { v8f   v; float f[8]; };

static __device__ __forceinline__ v8f wmma_bf16(v16bf a, v16bf b, v8f c) {
    // D = A(16x32 bf16) * B(32x16 bf16) + C(16x16 f32)
    return __builtin_amdgcn_wmma_f32_16x16x32_bf16(
        /*neg_a=*/false, a, /*neg_b=*/false, b,
        /*c_mod=*/(short)0, c, /*reuse_a=*/false, /*reuse_b=*/false);
}

// Native bf16 converts: clang lowers fptrunc pairs to v_cvt_pk_bf16_f32
static __device__ __forceinline__ unsigned short f2bf(float f) {
    return __builtin_bit_cast(unsigned short, (__bf16)f);
}
static __device__ __forceinline__ float bf2f(unsigned short h) {
    return (float)__builtin_bit_cast(__bf16, h);
}
static __device__ __forceinline__ unsigned pack2(float a, float b) {
    v2bf r = { (__bf16)a, (__bf16)b };   // fuses into one packed convert
    return __builtin_bit_cast(unsigned, r);
}

// ---------------------------------------------------------------------------
// CDNA5 async global->LDS DMA path (GLOBAL_LOAD_ASYNC_TO_LDS_B128, ASYNCcnt)
// ---------------------------------------------------------------------------
#if __has_builtin(__builtin_amdgcn_global_load_async_to_lds_b128) && \
    __has_builtin(__builtin_amdgcn_s_wait_asynccnt)
#define USE_ASYNC_LDS 1
#else
#define USE_ASYNC_LDS 0
#endif

#define ASG __attribute__((address_space(1)))
#define ASL __attribute__((address_space(3)))
typedef ASG v4i* gv4ip;   // global int4*  (async src)
typedef ASL v4i* lv4ip;   // LDS int4*     (async dst)

// ---------------------------------------------------------------------------
// Problem constants
// ---------------------------------------------------------------------------
#define DIM   256
#define HEADS 8
#define CPH   32            // channels per head
#define WS    8
#define NB    2             // batch
#define IMH   192
#define IMW   192
#define HW    (IMH * IMW)   // 36864 pixels per channel plane
#define NPIX  HW            // GEMM N per batch

// ---------------------------------------------------------------------------
// Kernel 1: tiled GEMM  Y[b][m][n] = sum_k W[rowOff+m][k] * X[b][k][n] + bias[rowOff+m]
// 128x128 tile, K-step 32, bf16 WMMA 16x16x32, fp32 accumulate.
// Tiles staged in LDS as fp32 via async B128 DMA (no VGPR round trip);
// fp32->bf16 conversion fused into per-wave fragment packing.
// 256 threads = 8 waves arranged 4(M) x 2(N); each wave: 32x64 = 2x4 frags.
// ---------------------------------------------------------------------------
__global__ __launch_bounds__(256)
void gemm_bias_kernel(const float* __restrict__ W, const float* __restrict__ bias,
                      int wRowOff, int N, int K,
                      const float* __restrict__ X, long strideXb,
                      float* __restrict__ Y, long strideYb)
{
    const int nb = blockIdx.x, mb = blockIdx.y, b = blockIdx.z;
    const float* Xb = X + (long)b * strideXb;
    float*       Yb = Y + (long)b * strideYb;
    const int rowBase = mb * 128;
    const int n0      = nb * 128;

    // A: [m][k] fp32, 36-float row stride -> 144B rows (16B aligned for B128
    //    writes) and 36*i mod 64 distinct for 16 consecutive rows (no bank
    //    conflicts on column-wise fragment reads).
    // B: [k][n] fp32, kept in memory order (async copy is a straight DMA);
    //    fragment reads are row-wise across lanes -> conflict-free unpadded.
    __shared__ __attribute__((aligned(16))) float Af[128][36];
    __shared__ __attribute__((aligned(16))) float Bf[32][128];

    const int t = threadIdx.x;
    const int lane = t & 31, wave = t >> 5;
    const int waveM = wave >> 1, waveN = wave & 1;
    const int hi = lane >> 4, l15 = lane & 15;

    FragF acc[2][4];
#pragma unroll
    for (int mt = 0; mt < 2; ++mt)
#pragma unroll
        for (int nt = 0; nt < 4; ++nt)
#pragma unroll
            for (int r = 0; r < 8; ++r) acc[mt][nt].f[r] = 0.0f;

    for (int k0 = 0; k0 < K; k0 += 32) {
#if USE_ASYNC_LDS
        // --- DMA A tile: 128x32 fp32 = 1024 x 16B chunks, 4 per thread ---
#pragma unroll
        for (int i = 0; i < 4; ++i) {
            int cidx = t + 256 * i;
            int m = cidx >> 3, kq = cidx & 7;          // 8 chunks per row
            __builtin_amdgcn_global_load_async_to_lds_b128(
                (gv4ip)&W[(long)(wRowOff + rowBase + m) * K + k0 + kq * 4],
                (lv4ip)&Af[m][kq * 4], 0, 0);
        }
        // --- DMA B tile: 32x128 fp32 = 1024 x 16B chunks, 4 per thread ---
#pragma unroll
        for (int i = 0; i < 4; ++i) {
            int cidx = t + 256 * i;
            int kk = cidx >> 5, nq = cidx & 31;        // 32 chunks per k-row
            __builtin_amdgcn_global_load_async_to_lds_b128(
                (gv4ip)&Xb[(long)(k0 + kk) * N + n0 + nq * 4],
                (lv4ip)&Bf[kk][nq * 4], 0, 0);
        }
        __builtin_amdgcn_s_wait_asynccnt(0);
#else
        // --- fallback staging through VGPRs ---
#pragma unroll
        for (int i = 0; i < 16; ++i) {
            int idx = t + 256 * i;
            int m = idx >> 5, c = idx & 31;
            Af[m][c] = W[(long)(wRowOff + rowBase + m) * K + (k0 + c)];
        }
#pragma unroll
        for (int i = 0; i < 16; ++i) {
            int idx = t + 256 * i;
            int kk = idx >> 7, n = idx & 127;
            Bf[kk][n] = Xb[(long)(k0 + kk) * N + (n0 + n)];
        }
#endif
        // prefetch next K tile into the cache hierarchy (global_prefetch_b8)
        if (k0 + 32 < K) {
            __builtin_prefetch(&Xb[(long)(k0 + 32 + (t >> 7)) * N + n0 + (t & 127)], 0, 3);
            __builtin_prefetch(&W[(long)(wRowOff + rowBase + (t >> 1)) * K + k0 + 32], 0, 3);
        }
        __syncthreads();

        // --- pack fragments (ISA 7.12.2 layouts), fp32 -> bf16 fused here ---
        FragBF afr[2], bfr[4];
#pragma unroll
        for (int mt = 0; mt < 2; ++mt) {
            int row = waveM * 32 + mt * 16 + l15;
            const float* rp = Af[row];
#pragma unroll
            for (int v = 0; v < 8; ++v) {
                // A: kLocal = 8*hi + 16*(v>>2) + 2*(v&3), pair (kl, kl+1)
                int kl = 8 * hi + 16 * (v >> 2) + 2 * (v & 3);
                afr[mt].u[v] = pack2(rp[kl], rp[kl + 1]);
            }
        }
#pragma unroll
        for (int nt = 0; nt < 4; ++nt) {
            int n = waveN * 64 + nt * 16 + l15;
#pragma unroll
            for (int v = 0; v < 8; ++v) {
                // B: K = 8*hi + 16*(v>>2) + 2*(v&3) + {0,1}; lanes read consecutive n -> no conflicts
                int k = 8 * hi + 16 * (v >> 2) + 2 * (v & 3);
                bfr[nt].u[v] = pack2(Bf[k][n], Bf[k + 1][n]);
            }
        }
#pragma unroll
        for (int mt = 0; mt < 2; ++mt)
#pragma unroll
            for (int nt = 0; nt < 4; ++nt)
                acc[mt][nt].v = wmma_bf16(afr[mt].v, bfr[nt].v, acc[mt][nt].v);
        __syncthreads();
    }

    // --- epilogue: bias + fp32 store (C layout: row = r + 8*hi, col = l15) ---
#pragma unroll
    for (int mt = 0; mt < 2; ++mt)
#pragma unroll
        for (int nt = 0; nt < 4; ++nt)
#pragma unroll
            for (int r = 0; r < 8; ++r) {
                int m = rowBase + waveM * 32 + mt * 16 + r + 8 * hi;
                int n = n0 + waveN * 64 + nt * 16 + l15;
                Yb[(long)m * N + n] = acc[mt][nt].f[r] + bias[wRowOff + m];
            }
}

// ---------------------------------------------------------------------------
// Kernel 2: fused dwconv3x3 + L2norm + channel attention per (window, head).
// One wave (32 threads) per block. grid = (24*24 windows, B, HEADS).
// S = qn(32x64) . k2n^T : 2x2 frags x 2 K-steps = 8 WMMA
// O = softmax(S*temp)(32x32) . v2(32x64) : 2x4 frags = 8 WMMA
// ---------------------------------------------------------------------------
__global__ __launch_bounds__(32)
void attn_win_kernel(const float* __restrict__ bufQ,   // [B][256][H][W]  (q channels, post conv1x1+bias)
                     const float* __restrict__ bufKV,  // [B][512][H][W]  (k2 then v2 channels)
                     const float* __restrict__ w_dw,   const float* __restrict__ b_dw,
                     const float* __restrict__ w_dw2,  const float* __restrict__ b_dw2,
                     const float* __restrict__ temp,
                     float* __restrict__ bufO)         // [B][256][H][W]
{
    const int win  = blockIdx.x;
    const int b    = blockIdx.y;
    const int head = blockIdx.z;
    const int wy = win / (IMW / WS), wx = win % (IMW / WS);
    const int y0 = wy * WS, x0 = wx * WS;
    const int lane = threadIdx.x;
    const int hi = lane >> 4, l15 = lane & 15;

    // bf16 tiles [channel][pixel]; stride 66 (33 dwords) keeps u32 access aligned
    __shared__ __attribute__((aligned(16))) unsigned short qT[CPH][66];
    __shared__ __attribute__((aligned(16))) unsigned short kT[CPH][66];
    __shared__ __attribute__((aligned(16))) unsigned short vT[CPH][66];
    __shared__ __attribute__((aligned(16))) unsigned short pT[CPH][34]; // softmax probs
    __shared__ float invq[CPH], invk[CPH];

    // ---------------- fused depthwise 3x3 (zeros padding) into LDS ------------
#pragma unroll
    for (int m = 0; m < 3; ++m) {
        const float* src; const float* wdw; const float* bdw;
        unsigned short (*dst)[66];
        if (m == 0) {        // q  <- bufQ channels [head*32, +32), dw rows 0+head*32
            src = bufQ + ((long)b * DIM + head * CPH) * HW;
            wdw = w_dw + (head * CPH) * 9;       bdw = b_dw + head * CPH;       dst = qT;
        } else if (m == 1) { // k2 <- bufKV rows [head*32), dw2 rows 256+head*32
            src = bufKV + ((long)b * 2 * DIM + head * CPH) * HW;
            wdw = w_dw2 + (DIM + head * CPH) * 9; bdw = b_dw2 + DIM + head * CPH; dst = kT;
        } else {             // v2 <- bufKV rows [256+head*32), dw2 rows 512+head*32
            src = bufKV + ((long)b * 2 * DIM + DIM + head * CPH) * HW;
            wdw = w_dw2 + (2 * DIM + head * CPH) * 9; bdw = b_dw2 + 2 * DIM + head * CPH; dst = vT;
        }
        for (int c = 0; c < CPH; ++c) {
            const float* sc = src + (long)c * HW;
            float w9[9];
#pragma unroll
            for (int j = 0; j < 9; ++j) w9[j] = wdw[c * 9 + j]; // uniform -> scalar loads
            float bb = bdw[c];
#pragma unroll
            for (int half = 0; half < 2; ++half) {
                int p  = half * 32 + lane;
                int py = p >> 3, px = p & 7;
                int gy = y0 + py, gx = x0 + px;
                float acc = bb;
#pragma unroll
                for (int dy = 0; dy < 3; ++dy) {
                    int yy = gy + dy - 1;
                    if (yy < 0 || yy >= IMH) continue;
#pragma unroll
                    for (int dx = 0; dx < 3; ++dx) {
                        int xx = gx + dx - 1;
                        if (xx < 0 || xx >= IMW) continue;
                        acc += w9[dy * 3 + dx] * sc[yy * IMW + xx];
                    }
                }
                dst[c][p] = f2bf(acc);
            }
        }
    }
    __syncthreads();

    // ---------------- per-channel L2 norms over the 64 window pixels ----------
    {
        int c = lane;
        float sq = 0.f, sk = 0.f;
        for (int p = 0; p < 64; ++p) {
            float a = bf2f(qT[c][p]); sq += a * a;
            float d = bf2f(kT[c][p]); sk += d * d;
        }
        invq[c] = 1.0f / fmaxf(sqrtf(sq), 1e-12f);
        invk[c] = 1.0f / fmaxf(sqrtf(sk), 1e-12f);
    }
    __syncthreads();

    // ---------------- S = qn . k2n^T  (M=32 qc, N=32 kc, K=64 pixels) ---------
    FragF s[2][2];
#pragma unroll
    for (int mt = 0; mt < 2; ++mt)
#pragma unroll
        for (int nt = 0; nt < 2; ++nt)
#pragma unroll
            for (int r = 0; r < 8; ++r) s[mt][nt].f[r] = 0.0f;

    for (int kk = 0; kk < 64; kk += 32) {
        FragBF aq[2], bk[2];
#pragma unroll
        for (int mt = 0; mt < 2; ++mt) {
            int row = mt * 16 + l15;
            float iq = invq[row];
#pragma unroll
            for (int v = 0; v < 8; ++v) {
                int kl = kk + 8 * hi + 16 * (v >> 2) + 2 * (v & 3);
                aq[mt].u[v] = pack2(bf2f(qT[row][kl]) * iq, bf2f(qT[row][kl + 1]) * iq);
            }
        }
#pragma unroll
        for (int nt = 0; nt < 2; ++nt) {
            int d = nt * 16 + l15;
            float ik = invk[d];
#pragma unroll
            for (int v = 0; v < 8; ++v) {
                int kl = kk + 8 * hi + 16 * (v >> 2) + 2 * (v & 3);
                bk[nt].u[v] = pack2(bf2f(kT[d][kl]) * ik, bf2f(kT[d][kl + 1]) * ik);
            }
        }
#pragma unroll
        for (int mt = 0; mt < 2; ++mt)
#pragma unroll
            for (int nt = 0; nt < 2; ++nt)
                s[mt][nt].v = wmma_bf16(aq[mt].v, bk[nt].v, s[mt][nt].v);
    }

    // ---------------- softmax over the 32 key-channel columns -----------------
    const float tscale = temp[head];
#pragma unroll
    for (int mt = 0; mt < 2; ++mt)
#pragma unroll
        for (int r = 0; r < 8; ++r) {
            float a0 = s[mt][0].f[r] * tscale;
            float a1 = s[mt][1].f[r] * tscale;
            float mx = fmaxf(a0, a1);
#pragma unroll
            for (int msk = 1; msk < 16; msk <<= 1)  // stays within 16-lane half (hi fixed)
                mx = fmaxf(mx, __shfl_xor(mx, msk, 32));
            float e0 = __expf(a0 - mx), e1 = __expf(a1 - mx);
            float sm = e0 + e1;
#pragma unroll
            for (int msk = 1; msk < 16; msk <<= 1)
                sm += __shfl_xor(sm, msk, 32);
            float inv = 1.0f / sm;
            int row = mt * 16 + r + 8 * hi;       // C-layout row
            pT[row][l15]      = f2bf(e0 * inv);   // columns 0..15
            pT[row][16 + l15] = f2bf(e1 * inv);   // columns 16..31
        }
    __syncthreads();

    // ---------------- O = P . v2   (M=32 qc, K=32 kc, N=64 pixels) ------------
    FragBF ap[2], bv[4];
#pragma unroll
    for (int mt = 0; mt < 2; ++mt) {
        int row = mt * 16 + l15;
        const unsigned* rp = (const unsigned*)pT[row]; // 17-dword stride, aligned
#pragma unroll
        for (int v = 0; v < 8; ++v)
            ap[mt].u[v] = rp[4 * hi + 8 * (v >> 2) + (v & 3)];
    }
#pragma unroll
    for (int nt = 0; nt < 4; ++nt) {
        int p = nt * 16 + l15;
#pragma unroll
        for (int v = 0; v < 8; ++v) {
            int k = 8 * hi + 16 * (v >> 2) + 2 * (v & 3);
            bv[nt].u[v] = pack2(bf2f(vT[k][p]), bf2f(vT[k + 1][p]));
        }
    }
    FragF o[2][4];
#pragma unroll
    for (int mt = 0; mt < 2; ++mt)
#pragma unroll
        for (int nt = 0; nt < 4; ++nt) {
#pragma unroll
            for (int r = 0; r < 8; ++r) o[mt][nt].f[r] = 0.0f;
            o[mt][nt].v = wmma_bf16(ap[mt].v, bv[nt].v, o[mt][nt].v);
        }

    // ---------------- window_merge store ---------------------------------------
    const long outBase = ((long)b * DIM + head * CPH) * HW;
#pragma unroll
    for (int mt = 0; mt < 2; ++mt)
#pragma unroll
        for (int nt = 0; nt < 4; ++nt)
#pragma unroll
            for (int r = 0; r < 8; ++r) {
                int m = mt * 16 + r + 8 * hi;
                int p = nt * 16 + l15;
                int py = p >> 3, px = p & 7;
                bufO[outBase + (long)m * HW + (y0 + py) * IMW + (x0 + px)] = o[mt][nt].f[r];
            }
}

// ---------------------------------------------------------------------------
// Launch: 3 GEMMs (q-conv, kv-conv, proj) + fused window attention.
// Only q of branch 1 and k,v of branch 2 are used by the reference -> skip
// the dead 2/3 + 1/3 of the qkv GEMMs.
// Workspace: bufQ 2*256*HW | bufKV 2*512*HW | bufO 2*256*HW  (fp32, ~302 MB)
// ---------------------------------------------------------------------------
extern "C" void kernel_launch(void* const* d_in, const int* in_sizes, int n_in,
                              void* d_out, int out_size, void* d_ws, size_t ws_size,
                              hipStream_t stream)
{
    const float* x      = (const float*)d_in[0];
    const float* x2     = (const float*)d_in[1];
    const float* w_qkv  = (const float*)d_in[2];
    const float* b_qkv  = (const float*)d_in[3];
    const float* w_dw   = (const float*)d_in[4];
    const float* b_dw   = (const float*)d_in[5];
    const float* w_qkv2 = (const float*)d_in[6];
    const float* b_qkv2 = (const float*)d_in[7];
    const float* w_dw2  = (const float*)d_in[8];
    const float* b_dw2  = (const float*)d_in[9];
    const float* temp   = (const float*)d_in[10];
    const float* w_proj = (const float*)d_in[11];
    const float* b_proj = (const float*)d_in[12];
    float* out = (float*)d_out;

    float* bufQ  = (float*)d_ws;                          // [2][256][HW]
    float* bufKV = bufQ  + (size_t)NB * DIM * HW;         // [2][512][HW]
    float* bufO  = bufKV + (size_t)NB * 2 * DIM * HW;     // [2][256][HW]

    // conv1x1 (q only): rows 0..255 of w_qkv
    {
        dim3 grid(NPIX / 128, DIM / 128, NB);
        gemm_bias_kernel<<<grid, 256, 0, stream>>>(
            w_qkv, b_qkv, /*rowOff=*/0, NPIX, DIM,
            x, (long)DIM * HW, bufQ, (long)DIM * HW);
    }
    // conv1x1 (k2,v2 only): rows 256..767 of w_qkv2
    {
        dim3 grid(NPIX / 128, (2 * DIM) / 128, NB);
        gemm_bias_kernel<<<grid, 256, 0, stream>>>(
            w_qkv2, b_qkv2, /*rowOff=*/DIM, NPIX, DIM,
            x2, (long)DIM * HW, bufKV, (long)2 * DIM * HW);
    }
    // fused dwconv3x3 + l2norm + channel attention + window merge
    {
        dim3 grid((IMH / WS) * (IMW / WS), NB, HEADS);
        attn_win_kernel<<<grid, 32, 0, stream>>>(
            bufQ, bufKV, w_dw, b_dw, w_dw2, b_dw2, temp, bufO);
    }
    // projection conv1x1
    {
        dim3 grid(NPIX / 128, DIM / 128, NB);
        gemm_bias_kernel<<<grid, 256, 0, stream>>>(
            w_proj, b_proj, /*rowOff=*/0, NPIX, DIM,
            bufO, (long)DIM * HW, out, (long)DIM * HW);
    }
}
